// Attention_9603546874373
// MI455X (gfx1250) — matrix operand order repaired and checked by reading, not compiled, not run on hardware
//
#include <hip/hip_runtime.h>
#include <hip/hip_bf16.h>

typedef _Float16 half_t;
typedef __attribute__((ext_vector_type(16))) _Float16 v16h;
typedef __attribute__((ext_vector_type(8)))  _Float16 v8h;
typedef __attribute__((ext_vector_type(8)))  float    v8f;

#define DIM    768
#define NHEAD  12
#define HDIM   64
#define BATCH  8
#define SEQ    1024
#define ROWS   (BATCH * SEQ)      // 8192
#define QK_SCALE 0.125f           // 64^-0.5

// ---------------------------------------------------------------------------
// CDNA5 async global->LDS copy (ASYNCcnt-tracked, bypasses VGPRs).
// LDS operand is the byte offset from the workgroup LDS base; for a generic
// pointer to a __shared__ object that is exactly the low 32 address bits
// (LDS aperture keeps the offset in addr[31:0]).
// ---------------------------------------------------------------------------
__device__ __forceinline__ void async_copy_b128(void* lds_dst, const void* gsrc) {
  asm volatile("global_load_async_to_lds_b128 %0, %1, off"
               :: "v"((unsigned int)(unsigned long long)lds_dst), "v"(gsrc)
               : "memory");
}
__device__ __forceinline__ void wait_async0() {
  asm volatile("s_wait_asynccnt 0x0" ::: "memory");
}

// ---------------------------------------------------------------------------
// WMMA fragment helpers (wave32; A/B use contiguous k order k = 16*(lane/16) + i)
// ---------------------------------------------------------------------------

// A (16x32 f16, MxK): lanes 0-15 row M=lane, K=0..15;
// lanes 16-31 row M=lane-16, K=16..31.
__device__ __forceinline__ v16h frag_a(const half_t* s, int ld, int row0, int col0) {
  int lane = threadIdx.x & 31;
  int g = lane >> 4, m = lane & 15;
  const half_t* p = s + (row0 + m) * ld + col0 + g * 16;
  v16h a;
#pragma unroll
  for (int h = 0; h < 16; ++h) a[h] = p[h];
  return a;
}

// B (32x16 f16, KxN) with B[k][n] = src[n0+n][kb+k]  (src rows are output
// columns). Lanes 0-15: N=lane, K=0..15; lanes 16-31: N=lane-16, K=16..31.
__device__ __forceinline__ v16h frag_bt(const half_t* s, int ld, int n0, int kb) {
  int lane = threadIdx.x & 31;
  int g = lane >> 4, n = lane & 15;
  const half_t* p = s + (n0 + n) * ld + kb + g * 16;
  v16h b;
#pragma unroll
  for (int h = 0; h < 16; ++h) b[h] = p[h];
  return b;
}

// B (32x16 f16, KxN) with B[k][n] = src[k0+k][n0+n]  (row-major KxN source).
__device__ __forceinline__ v16h frag_bn(const half_t* s, int ld, int k0, int n0) {
  int lane = threadIdx.x & 31;
  int g = lane >> 4, n = lane & 15;
  v16h b;
#pragma unroll
  for (int h = 0; h < 16; ++h) b[h] = s[(k0 + g * 16 + h) * ld + n0 + n];
  return b;
}

// C/D (16x16 f32): VGPR r -> (M = r + 8*(lane>=16), N = lane%16)
__device__ __forceinline__ void store_c(float* s, int ld, int row0, int col0, v8f c) {
  int lane = threadIdx.x & 31;
  int g = lane >> 4, n = lane & 15;
#pragma unroll
  for (int r = 0; r < 8; ++r) s[(row0 + r + g * 8) * ld + col0 + n] = c[r];
}

__device__ __forceinline__ v8f load_c(const float* s, int ld, int row0, int col0) {
  int lane = threadIdx.x & 31;
  int g = lane >> 4, n = lane & 15;
  v8f c;
#pragma unroll
  for (int r = 0; r < 8; ++r) c[r] = s[(row0 + r + g * 8) * ld + col0 + n];
  return c;
}

__device__ __forceinline__ v8f wmma16(v16h a, v16h b, v8f c) {
  return __builtin_amdgcn_wmma_f32_16x16x32_f16(false, a, false, b, (short)0, c,
                                                false, false);
}

// ---------------------------------------------------------------------------
// f32 -> f16 conversion
// ---------------------------------------------------------------------------
__global__ void cvt_f32_f16(const float* __restrict__ src, half_t* __restrict__ dst,
                            int n) {
  int i = blockIdx.x * blockDim.x + threadIdx.x;
  if (i < n) dst[i] = (half_t)src[i];
}

// ---------------------------------------------------------------------------
// GEMM: C[M x N] = A[M x 768] * B[N x 768]^T, f16 in, f32 acc.
// mode 0: qkv — add [q_bias, 0, v_bias], scale Q, scatter into (b,h,n,d) f16.
// mode 1: proj — add b_proj, write f32 to out.
// One 256-thread block computes a 64x64 C tile (8 waves x 2 WMMA tiles).
// ---------------------------------------------------------------------------
__global__ __launch_bounds__(256)
void gemm_kernel(const half_t* __restrict__ A, const half_t* __restrict__ B,
                 const float* __restrict__ bias0, const float* __restrict__ bias2,
                 half_t* __restrict__ qb, half_t* __restrict__ kb,
                 half_t* __restrict__ vb, float* __restrict__ outf, int mode) {
  __shared__ half_t sA[64 * 40];   // 64 rows x 32 K, ld 40 (16B aligned rows)
  __shared__ half_t sB[64 * 40];

  const int m0 = blockIdx.x * 64;
  const int n0 = blockIdx.y * 64;
  const int t  = threadIdx.x;
  const int w  = t >> 5;

  const v8f zero = {0.f, 0.f, 0.f, 0.f, 0.f, 0.f, 0.f, 0.f};
  v8f acc[2];
  acc[0] = zero;
  acc[1] = zero;

  const int idx = t * 8;
  const int lr = idx >> 5;      // 0..63
  const int lc = idx & 31;      // multiple of 8

  for (int kt = 0; kt < DIM / 32; ++kt) {
    __syncthreads();
    // async-stage 64x32 tiles of A and B (16 bytes per thread each, no VGPR
    // round trip; tracked on ASYNCcnt)
    async_copy_b128(&sA[lr * 40 + lc], &A[(m0 + lr) * DIM + kt * 32 + lc]);
    async_copy_b128(&sB[lr * 40 + lc], &B[(n0 + lr) * DIM + kt * 32 + lc]);
    wait_async0();
    __syncthreads();
#pragma unroll
    for (int j = 0; j < 2; ++j) {
      int ti = w * 2 + j;
      int mi = ti >> 2, ni = ti & 3;
      v16h a  = frag_a(sA, 40, mi * 16, 0);
      v16h bf = frag_bt(sB, 40, ni * 16, 0);
      acc[j] = wmma16(a, bf, acc[j]);
    }
  }

  // epilogue
  const int lane = t & 31;
  const int g = lane >> 4, nn = lane & 15;
#pragma unroll
  for (int j = 0; j < 2; ++j) {
    int ti = w * 2 + j;
    int mi = ti >> 2, ni = ti & 3;
#pragma unroll
    for (int r = 0; r < 8; ++r) {
      int row = m0 + mi * 16 + r + g * 8;
      int col = n0 + ni * 16 + nn;
      float v = acc[j][r];
      if (mode == 0) {
        int which = col / DIM;           // 0=q 1=k 2=v (uniform per tile)
        int cc = col - which * DIM;      // 0..767
        int h = cc >> 6, d = cc & 63;
        if (which == 0) v += bias0[cc];
        else if (which == 2) v += bias2[cc];
        int bi = row >> 10, nq = row & 1023;
        size_t o = ((size_t)(bi * NHEAD + h) * SEQ + nq) * HDIM + d;
        if (which == 0)      qb[o] = (half_t)(v * QK_SCALE);
        else if (which == 1) kb[o] = (half_t)v;
        else                 vb[o] = (half_t)v;
      } else {
        outf[(size_t)row * DIM + col] = v + bias0[col];
      }
    }
  }
}

// ---------------------------------------------------------------------------
// Flash attention: one block = one (b*h, 64-query block).
// Q pre-scaled; online softmax over 16 key tiles of 64.
// ---------------------------------------------------------------------------
__global__ __launch_bounds__(256)
void attn_kernel(const half_t* __restrict__ Q, const half_t* __restrict__ K,
                 const half_t* __restrict__ V, half_t* __restrict__ O) {
  __shared__ half_t sQ[64 * 72];
  __shared__ half_t sK[64 * 72];
  __shared__ half_t sV[64 * 72];
  __shared__ float  sS[64 * 66];   // scores; reused in-place as f16 P (ld 132)
  __shared__ float  sO[64 * 66];   // f32 output accumulator
  __shared__ float  sm[64], sl[64], sal[64];
  half_t* sP = (half_t*)sS;

  const int bh = blockIdx.x;       // 0..95
  const int qb = blockIdx.y;       // 0..15
  const int t  = threadIdx.x;
  const int w  = t >> 5;

  const half_t* Qg = Q + (size_t)bh * SEQ * HDIM + (size_t)qb * 64 * HDIM;
  const half_t* Kg = K + (size_t)bh * SEQ * HDIM;
  const half_t* Vg = V + (size_t)bh * SEQ * HDIM;

  // async-load Q tile (64x64 halves) + init state
  for (int base = t * 8; base < 64 * 64; base += 256 * 8) {
    int r = base >> 6, c = base & 63;
    async_copy_b128(&sQ[r * 72 + c], &Qg[r * 64 + c]);
  }
  for (int i = t; i < 64 * 66; i += 256) sO[i] = 0.f;
  if (t < 64) { sm[t] = -3.0e38f; sl[t] = 0.f; }
  wait_async0();

  const v8f zero = {0.f, 0.f, 0.f, 0.f, 0.f, 0.f, 0.f, 0.f};

  for (int kt = 0; kt < SEQ / 64; ++kt) {
    __syncthreads();   // prior P@V done before overwriting K/V
    for (int base = t * 8; base < 64 * 64; base += 256 * 8) {
      int r = base >> 6, c = base & 63;
      async_copy_b128(&sK[r * 72 + c], &Kg[(kt * 64 + r) * 64 + c]);
      async_copy_b128(&sV[r * 72 + c], &Vg[(kt * 64 + r) * 64 + c]);
    }
    wait_async0();
    __syncthreads();

    // S = Q * K^T  (16 tiles of 16x16, 2 per wave, 2 K-slices each)
#pragma unroll
    for (int j = 0; j < 2; ++j) {
      int ti = w * 2 + j;
      int qi = ti >> 2, ki = ti & 3;
      v8f c = zero;
      c = wmma16(frag_a(sQ, 72, qi * 16, 0),  frag_bt(sK, 72, ki * 16, 0),  c);
      c = wmma16(frag_a(sQ, 72, qi * 16, 32), frag_bt(sK, 72, ki * 16, 32), c);
      store_c(sS, 66, qi * 16, ki * 16, c);
    }
    __syncthreads();

    // online softmax: thread r owns row r; P written f16 in-place (front-to-
    // back, so the f16 write at slot c/2 never clobbers an unread f32 at c).
    if (t < 64) {
      float mo = sm[t];
      float mt = mo;
      for (int c = 0; c < 64; ++c) mt = fmaxf(mt, sS[t * 66 + c]);
      float al = __expf(mo - mt);
      float sum = 0.f;
      for (int c = 0; c < 64; ++c) {
        float p = __expf(sS[t * 66 + c] - mt);
        sP[t * 132 + c] = (half_t)p;
        sum += p;
      }
      sl[t] = sl[t] * al + sum;
      sm[t] = mt;
      sal[t] = al;
    }
    __syncthreads();

    // rescale O accumulator
    for (int i = t; i < 64 * 64; i += 256) {
      int r = i >> 6;
      sO[r * 66 + (i & 63)] *= sal[r];
    }
    __syncthreads();

    // O += P * V
#pragma unroll
    for (int j = 0; j < 2; ++j) {
      int ti = w * 2 + j;
      int qi = ti >> 2, di = ti & 3;
      v8f c = load_c(sO, 66, qi * 16, di * 16);
      c = wmma16(frag_a(sP, 132, qi * 16, 0),  frag_bn(sV, 72, 0,  di * 16), c);
      c = wmma16(frag_a(sP, 132, qi * 16, 32), frag_bn(sV, 72, 32, di * 16), c);
      store_c(sO, 66, qi * 16, di * 16, c);
    }
  }
  __syncthreads();

  // normalize and write to (b, n, h*64+d) f16 for the proj GEMM
  const int b = bh / NHEAD, h = bh % NHEAD;
  for (int i = t; i < 64 * 64; i += 256) {
    int r = i >> 6, d = i & 63;
    float v = sO[r * 66 + d] / sl[r];
    int n = qb * 64 + r;
    O[(size_t)(b * SEQ + n) * DIM + h * HDIM + d] = (half_t)v;
  }
}

// ---------------------------------------------------------------------------
// Host launch
// ---------------------------------------------------------------------------
extern "C" void kernel_launch(void* const* d_in, const int* in_sizes, int n_in,
                              void* d_out, int out_size, void* d_ws, size_t ws_size,
                              hipStream_t stream) {
  const float* x      = (const float*)d_in[0];
  const float* w_qkv  = (const float*)d_in[1];
  const float* q_bias = (const float*)d_in[2];
  const float* v_bias = (const float*)d_in[3];
  const float* w_proj = (const float*)d_in[4];
  const float* b_proj = (const float*)d_in[5];
  float* out = (float*)d_out;

  char* ws = (char*)d_ws;
  const size_t szXH  = (size_t)ROWS * DIM * sizeof(half_t);        // 12.6 MB
  const size_t szWQ  = (size_t)3 * DIM * DIM * sizeof(half_t);     //  3.5 MB
  const size_t szWP  = (size_t)DIM * DIM * sizeof(half_t);         //  1.2 MB
  const size_t szHD  = (size_t)BATCH * NHEAD * SEQ * HDIM * sizeof(half_t);

  half_t* xh   = (half_t*)ws;               ws += szXH;
  half_t* wqh  = (half_t*)ws;               ws += szWQ;
  half_t* wph  = (half_t*)ws;               ws += szWP;
  half_t* qbuf = (half_t*)ws;               ws += szHD;
  half_t* kbuf = (half_t*)ws;               ws += szHD;
  half_t* vbuf = (half_t*)ws;               ws += szHD;
  half_t* atth = (half_t*)ws;               ws += szXH;
  (void)ws_size; (void)in_sizes; (void)n_in; (void)out_size;

  // f32 -> f16 conversions
  {
    int n1 = ROWS * DIM;
    cvt_f32_f16<<<(n1 + 255) / 256, 256, 0, stream>>>(x, xh, n1);
    int n2 = 3 * DIM * DIM;
    cvt_f32_f16<<<(n2 + 255) / 256, 256, 0, stream>>>(w_qkv, wqh, n2);
    int n3 = DIM * DIM;
    cvt_f32_f16<<<(n3 + 255) / 256, 256, 0, stream>>>(w_proj, wph, n3);
  }

  // QKV projection: (8192 x 2304) = xh * w_qkv^T, scatter into q/k/v
  {
    dim3 grid(ROWS / 64, (3 * DIM) / 64);
    gemm_kernel<<<grid, 256, 0, stream>>>(xh, wqh, q_bias, v_bias,
                                          qbuf, kbuf, vbuf, nullptr, 0);
  }

  // fused attention: 96 heads x 16 query blocks
  {
    dim3 grid(BATCH * NHEAD, SEQ / 64);
    attn_kernel<<<grid, 256, 0, stream>>>(qbuf, kbuf, vbuf, atth);
  }

  // output projection: (8192 x 768) = atth * w_proj^T + b_proj -> f32 out
  {
    dim3 grid(ROWS / 64, DIM / 64);
    gemm_kernel<<<grid, 256, 0, stream>>>(atth, wph, b_proj, nullptr,
                                          nullptr, nullptr, nullptr, out, 1);
  }
}
